// TabularMambaRegressor_38225208934428
// MI455X (gfx1250) — matrix operand order repaired and checked by reading, not compiled, not run on hardware
//
#include <hip/hip_runtime.h>
#include <math.h>
#include <stdint.h>

typedef _Float16 half8 __attribute__((ext_vector_type(8)));
typedef _Float16 v16h  __attribute__((ext_vector_type(16)));
typedef float    v8f   __attribute__((ext_vector_type(8)));

#define B_   128
#define L_   128
#define DM   256
#define DI   512
#define NS   16
#define BL   (B_ * L_)          // 16384
#define XPD  48                 // DT_RANK + 2*D_STATE

// ---------------------------------------------------------------------------
// CDNA5 async global->LDS copy (16B per lane, tracked by ASYNCcnt)
// ---------------------------------------------------------------------------
__device__ __forceinline__ void async_b128(uint32_t lds_off, const _Float16* g)
{
  asm volatile("global_load_async_to_lds_b128 %0, %1, off"
               :: "v"(lds_off), "v"(g) : "memory");
}
template <int CNT>
__device__ __forceinline__ void wait_async()
{
  asm volatile("s_wait_asynccnt %0" :: "n"(CNT) : "memory");
}

// ---------------------------------------------------------------------------
// weight conversion: f32 -> f16 (straight copy)
// ---------------------------------------------------------------------------
__global__ __launch_bounds__(256) void k_cvt_f16(
    const float* __restrict__ s, _Float16* __restrict__ d)
{
  int i = blockIdx.x * 256 + threadIdx.x;
  d[i] = (_Float16)s[i];
}

// x_proj_w: [2,48,512] f32 -> [2,64,512] f16, rows 48..63 zeroed (pad for tiles)
__global__ __launch_bounds__(256) void k_cvt_pad_wx(
    const float* __restrict__ s, _Float16* __restrict__ d)
{
  int i  = blockIdx.x * 256 + threadIdx.x;   // 2*64*512
  int c  = i & 511;
  int r  = (i >> 9) & 63;
  int ly = i >> 15;
  d[i] = (r < XPD) ? (_Float16)s[((size_t)ly * XPD + r) * DI + c] : (_Float16)0.0f;
}

// ---------------------------------------------------------------------------
// scalar projection: x[b,l,d] = x_num[b,l]*w[d] + bias[d]   (f16 out)
// ---------------------------------------------------------------------------
__global__ __launch_bounds__(256) void k_scalar_proj(
    const float* __restrict__ xn, const float* __restrict__ w,
    const float* __restrict__ bias, _Float16* __restrict__ x)
{
  int idx = blockIdx.x * 256 + threadIdx.x;     // BL*DM
  int d   = idx & (DM - 1);
  int bl  = idx >> 8;
  x[idx] = (_Float16)(xn[bl] * w[d] + bias[d]);
}

// ---------------------------------------------------------------------------
// WMMA GEMM, async double-buffered:  C[M,N](f16) = A[M,K](f16) * W[N,K]^T(f16)
// Preconditions: K % 32 == 0; W has rows padded to a multiple of 64.
// Block: 256 thr = 8 waves; tile 128(M)x64(N); wave 32x32 = 2x2 WMMA accs.
// Tiles move via global_load_async_to_lds_b128 (3 issues/thread/K-step).
// ---------------------------------------------------------------------------
__global__ __launch_bounds__(256) void k_gemm_wmma(
    const _Float16* __restrict__ A, int lda,
    const _Float16* __restrict__ W,
    _Float16* __restrict__ C,
    int M, int N, int K)
{
  __shared__ __align__(16) _Float16 As[2][128][32];   // 2 x 8 KB
  __shared__ __align__(16) _Float16 Ws[2][64][32];    // 2 x 4 KB

  const int tid  = threadIdx.x;
  const int lane = tid & 31;
  const int wave = tid >> 5;
  const int l16  = lane & 15;
  const int hsel = (lane >> 4) & 1;
  const int m0   = blockIdx.y * 128;
  const int n0   = blockIdx.x * 64;
  const int wm   = (wave & 3) * 32;
  const int wn   = (wave >> 2) * 32;
  const int nk   = K >> 5;

  // 16B-chunk decomposition: chunk e -> row e>>2, col (e&3)*8 halves
  const int ar0 = tid >> 2,         ac0 = (tid & 3) * 8;          // A chunk tid
  const int ar1 = (tid + 256) >> 2, ac1 = ac0;                    // A chunk tid+256
  const int wr  = tid >> 2,         wc  = ac0;                    // W chunk tid

  auto prefetch = [&](int kt, int buf) {
    const int k0 = kt << 5;
    async_b128((uint32_t)(uintptr_t)&As[buf][ar0][ac0],
               &A[(size_t)(m0 + ar0) * lda + k0 + ac0]);
    async_b128((uint32_t)(uintptr_t)&As[buf][ar1][ac1],
               &A[(size_t)(m0 + ar1) * lda + k0 + ac1]);
    async_b128((uint32_t)(uintptr_t)&Ws[buf][wr][wc],
               &W[(size_t)(n0 + wr) * K + k0 + wc]);
  };

  v8f acc[2][2] = {};
  prefetch(0, 0);

  for (int kt = 0; kt < nk; ++kt) {
    const int buf = kt & 1;
    if (kt + 1 < nk) { prefetch(kt + 1, buf ^ 1); wait_async<3>(); }
    else             { wait_async<0>(); }
    __syncthreads();   // tile kt fully in LDS for all waves

    v16h afrag[2], bfrag[2];
    #pragma unroll
    for (int mi = 0; mi < 2; ++mi) {
      // A layout: lane<16 -> K {kb..kb+7, kb+16..kb+23}, kb = hsel*8
      const half8* p = (const half8*)&As[buf][wm + mi * 16 + l16][hsel * 8];
      half8 lo = p[0];
      half8 hi = p[2];
      afrag[mi] = __builtin_shufflevector(lo, hi, 0,1,2,3,4,5,6,7,8,9,10,11,12,13,14,15);
    }
    #pragma unroll
    for (int ni = 0; ni < 2; ++ni) {
      // B layout: lane<16 -> K {kb..kb+7, kb+16..kb+23}, kb = hsel*8
      const half8* p = (const half8*)&Ws[buf][wn + ni * 16 + l16][hsel * 8];
      half8 lo = p[0];
      half8 hi = p[2];
      bfrag[ni] = __builtin_shufflevector(lo, hi, 0,1,2,3,4,5,6,7,8,9,10,11,12,13,14,15);
    }
    #pragma unroll
    for (int mi = 0; mi < 2; ++mi)
      #pragma unroll
      for (int ni = 0; ni < 2; ++ni)
        acc[mi][ni] = __builtin_amdgcn_wmma_f32_16x16x32_f16(
            false, afrag[mi], false, bfrag[ni], (short)0, acc[mi][ni], false, false);

    __syncthreads();   // all waves done reading buf before it is re-prefetched
  }

  #pragma unroll
  for (int mi = 0; mi < 2; ++mi)
    #pragma unroll
    for (int ni = 0; ni < 2; ++ni) {
      int colg = n0 + wn + ni * 16 + l16;
      if (colg < N) {
        #pragma unroll
        for (int j = 0; j < 8; ++j) {
          int rowg = m0 + wm + mi * 16 + j + hsel * 8;
          C[(size_t)rowg * N + colg] = (_Float16)acc[mi][ni][j];
        }
      }
    }
}

// ---------------------------------------------------------------------------
// causal depthwise conv1d (kernel 4) + bias + SiLU.   f16 in/out, f32 math.
// ---------------------------------------------------------------------------
__global__ __launch_bounds__(256) void k_conv_silu(
    const _Float16* __restrict__ xz, const float* __restrict__ cw,
    const float* __restrict__ cb, _Float16* __restrict__ xh)
{
  int idx = blockIdx.x * 256 + threadIdx.x;   // BL*DI
  int d   = idx & (DI - 1);
  int l   = (idx >> 9) & (L_ - 1);
  int bl0 = (idx >> 9) - l;                   // b*L
  float acc = cb[d];
  #pragma unroll
  for (int j = 0; j < 4; ++j) {
    int ls = l - 3 + j;
    if (ls >= 0)
      acc = fmaf(cw[d * 4 + j], (float)xz[(size_t)(bl0 + ls) * (2 * DI) + d], acc);
  }
  xh[idx] = (_Float16)(acc / (1.0f + expf(-acc)));   // silu
}

// ---------------------------------------------------------------------------
// selective scan, fused with: dt projection (16 MACs), softplus, +u*D_skip,
// *silu(z). One block per batch row, one thread per channel; dt/B/C rows for
// all L staged in LDS; 16-state recurrence in registers; y written in place.
// ---------------------------------------------------------------------------
__global__ __launch_bounds__(512) void k_scan(
    _Float16* __restrict__ xh,          // [B,L,DI]   u in, y out (in place)
    const _Float16* __restrict__ xz,    // [B,L,2*DI] (z at col offset DI)
    const _Float16* __restrict__ xdbl,  // [B,L,48]   (dt:0..15, Bm:16..31, C:32..47)
    const float* __restrict__ wdtm,     // [DI,16]  dt_proj_w
    const float* __restrict__ dtb,      // [DI]
    const float* __restrict__ alog,     // [DI,NS]
    const float* __restrict__ dsk)      // [DI]
{
  __shared__ float Dt[L_][NS];
  __shared__ float Bs[L_][NS];
  __shared__ float Cs[L_][NS];
  const int b = blockIdx.x;
  const int d = threadIdx.x;

  #pragma unroll
  for (int i = 0; i < (L_ * XPD) / 512; ++i) {      // 12 iters
    int e = threadIdx.x + i * 512;
    int l = e / XPD, c = e - l * XPD;
    float v = (float)xdbl[((size_t)b * L_ + l) * XPD + c];
    if (c < NS)           Dt[l][c]          = v;
    else if (c < 2 * NS)  Bs[l][c - NS]     = v;
    else                  Cs[l][c - 2 * NS] = v;
  }
  __syncthreads();

  float Ad[NS], h[NS], wdt[NS];
  #pragma unroll
  for (int n = 0; n < NS; ++n) {
    Ad[n]  = -expf(alog[d * NS + n]);
    wdt[n] = wdtm[d * NS + n];
    h[n]   = 0.0f;
  }
  const float bdt  = dtb[d];
  const float skip = dsk[d];

  for (int l = 0; l < L_; ++l) {
    size_t base = (size_t)b * L_ + l;
    float dr = bdt;
    #pragma unroll
    for (int r = 0; r < NS; ++r) dr = fmaf(Dt[l][r], wdt[r], dr);   // dt proj
    float delta = (dr > 20.0f) ? dr : log1pf(expf(dr));             // softplus
    float u = (float)xh[base * DI + d];
    float z = (float)xz[base * (2 * DI) + DI + d];
    float y = 0.0f;
    #pragma unroll
    for (int n = 0; n < NS; ++n) {
      float dA = expf(delta * Ad[n]);
      h[n] = fmaf(dA, h[n], delta * Bs[l][n] * u);
      y    = fmaf(h[n], Cs[l][n], y);
    }
    y = fmaf(u, skip, y);
    float sz = z / (1.0f + expf(-z));
    xh[base * DI + d] = (_Float16)(y * sz);
  }
}

// ---------------------------------------------------------------------------
// LayerNorm(DM) -> mean-pool over L -> Linear+GELU(exact) -> Linear -> [B] f32
// ---------------------------------------------------------------------------
__global__ __launch_bounds__(256) void k_head(
    const _Float16* __restrict__ x,   // [B,L,DM]
    const float* __restrict__ g, const float* __restrict__ beta,
    const float* __restrict__ h1w, const float* __restrict__ h1b,
    const float* __restrict__ h2w, const float* __restrict__ h2b,
    float* __restrict__ out)
{
  __shared__ float r1[DM];
  __shared__ float r2[DM];
  __shared__ float pool[DM];
  const int b = blockIdx.x;
  const int d = threadIdx.x;

  float pacc = 0.0f;
  for (int l = 0; l < L_; ++l) {
    float v = (float)x[((size_t)b * L_ + l) * DM + d];
    r1[d] = v; r2[d] = v * v;
    __syncthreads();
    for (int s = DM / 2; s > 0; s >>= 1) {
      if (d < s) { r1[d] += r1[d + s]; r2[d] += r2[d + s]; }
      __syncthreads();
    }
    float mu  = r1[0] * (1.0f / DM);
    float var = r2[0] * (1.0f / DM) - mu * mu;
    pacc += (v - mu) * rsqrtf(var + 1e-5f) * g[d] + beta[d];
    __syncthreads();
  }
  pool[d] = pacc * (1.0f / L_);
  __syncthreads();

  float hj = h1b[d];
  for (int k = 0; k < DM; ++k) hj = fmaf(pool[k], h1w[d * DM + k], hj);
  float ge = 0.5f * hj * (1.0f + erff(hj * 0.70710678118654752f));  // exact GELU
  r1[d] = ge * h2w[d];
  __syncthreads();
  for (int s = DM / 2; s > 0; s >>= 1) {
    if (d < s) r1[d] += r1[d + s];
    __syncthreads();
  }
  if (d == 0) out[b] = r1[0] + h2b[0];
}

// ---------------------------------------------------------------------------
extern "C" void kernel_launch(void* const* d_in, const int* in_sizes, int n_in,
                              void* d_out, int out_size, void* d_ws, size_t ws_size,
                              hipStream_t stream) {
  const float* x_num     = (const float*)d_in[0];
  const float* scalar_w  = (const float*)d_in[1];
  const float* scalar_b  = (const float*)d_in[2];
  const float* in_proj_w = (const float*)d_in[3];   // [2,1024,256]
  const float* conv_w    = (const float*)d_in[4];   // [2,512,4]
  const float* conv_b    = (const float*)d_in[5];   // [2,512]
  const float* x_proj_w  = (const float*)d_in[6];   // [2,48,512]
  const float* dt_proj_w = (const float*)d_in[7];   // [2,512,16]
  const float* dt_proj_b = (const float*)d_in[8];   // [2,512]
  const float* A_log     = (const float*)d_in[9];   // [2,512,16]
  const float* D_skip    = (const float*)d_in[10];  // [2,512]
  const float* out_proj  = (const float*)d_in[11];  // [2,256,512]
  const float* ln_g      = (const float*)d_in[12];
  const float* ln_b      = (const float*)d_in[13];
  const float* h1_w      = (const float*)d_in[14];
  const float* h1_b      = (const float*)d_in[15];
  const float* h2_w      = (const float*)d_in[16];
  const float* h2_b      = (const float*)d_in[17];
  float* out = (float*)d_out;

  // workspace layout (f16 halves)
  _Float16* xh_buf = (_Float16*)d_ws;                 // [BL, DM]
  _Float16* xz_h   = xh_buf + (size_t)BL * DM;        // [BL, 2*DI]
  _Float16* xhid_h = xz_h   + (size_t)BL * 2 * DI;    // [BL, DI]
  _Float16* xdbl_h = xhid_h + (size_t)BL * DI;        // [BL, 48]
  _Float16* Wi_h   = xdbl_h + (size_t)BL * XPD;       // [2,1024,256]
  _Float16* Wx_h   = Wi_h   + (size_t)2 * 2 * DI * DM;// [2,64,512] padded
  _Float16* Wo_h   = Wx_h   + (size_t)2 * 64 * DI;    // [2,256,512]

  // one-time (per launch) weight conversion to f16
  k_cvt_f16<<<(2 * 2 * DI * DM) / 256, 256, 0, stream>>>(in_proj_w, Wi_h);
  k_cvt_f16<<<(2 * DM * DI) / 256, 256, 0, stream>>>(out_proj, Wo_h);
  k_cvt_pad_wx<<<(2 * 64 * DI) / 256, 256, 0, stream>>>(x_proj_w, Wx_h);

  k_scalar_proj<<<(BL * DM) / 256, 256, 0, stream>>>(x_num, scalar_w, scalar_b, xh_buf);

  for (int i = 0; i < 2; ++i) {
    const _Float16* Wi = Wi_h + (size_t)i * 2 * DI * DM;
    const _Float16* Wx = Wx_h + (size_t)i * 64 * DI;
    const _Float16* Wo = Wo_h + (size_t)i * DM * DI;

    // xz = x @ Wi^T            [BL,1024]   K=256
    k_gemm_wmma<<<dim3((2 * DI) / 64, BL / 128), 256, 0, stream>>>(
        xh_buf, DM, Wi, xz_h, BL, 2 * DI, DM);
    // depthwise conv + silu -> xhid
    k_conv_silu<<<(BL * DI) / 256, 256, 0, stream>>>(
        xz_h, conv_w + (size_t)i * DI * 4, conv_b + (size_t)i * DI, xhid_h);
    // x_dbl = xh @ Wx^T        [BL,48]    K=512 (W rows padded to 64)
    k_gemm_wmma<<<dim3(1, BL / 128), 256, 0, stream>>>(
        xhid_h, DI, Wx, xdbl_h, BL, XPD, DI);
    // fused dt-proj + softplus + scan + D_skip + silu(z), in-place into xhid
    k_scan<<<B_, DI, 0, stream>>>(
        xhid_h, xz_h, xdbl_h,
        dt_proj_w + (size_t)i * DI * NS, dt_proj_b + (size_t)i * DI,
        A_log + (size_t)i * DI * NS, D_skip + (size_t)i * DI);
    // x = y @ Wo^T             [BL,256]   K=512
    k_gemm_wmma<<<dim3(DM / 64, BL / 128), 256, 0, stream>>>(
        xhid_h, DI, Wo, xh_buf, BL, DM, DI);
  }

  k_head<<<B_, DM, 0, stream>>>(xh_buf, ln_g, ln_b, h1_w, h1_b, h2_w, h2_b, out);

  (void)in_sizes; (void)n_in; (void)out_size; (void)ws_size;
}
